// DGCNNEncoder_41145786696430
// MI455X (gfx1250) — matrix operand order repaired and checked by reading, not compiled, not run on hardware
//
#include <hip/hip_runtime.h>
#include <hip/hip_bf16.h>
#include <math.h>

typedef __bf16 bf16;
typedef __attribute__((ext_vector_type(16))) __bf16 v16bf;
typedef __attribute__((ext_vector_type(8)))  float  v8f;

#define KNN   20
#define SLOPE 0.2f
#define NPTS  2048
#define BATCH 8

__device__ __forceinline__ v8f wmma_bf16(v16bf a, v16bf b, v8f c) {
    return __builtin_amdgcn_wmma_f32_16x16x32_bf16(
        false, a, false, b, (short)0, c, false, false);
}

// Load a 16-element bf16 fragment as two contiguous 16-byte runs.
// A-fragment: base = row + s*32 + 8*g, delta = 16 (K = 8g..8g+7, 16+8g..16+8g+7)
// B-fragment: base = row + s*32 + 8*g, delta = 16 (K = 8g..8g+7, 16+8g..16+8g+7)
__device__ __forceinline__ v16bf load_frag(const bf16* __restrict__ p, int delta) {
    union { uint4 u[2]; v16bf v; } un;
    un.u[0] = *(const uint4*)p;
    un.u[1] = *(const uint4*)(p + delta);
    return un.v;
}

// ---------------------------------------------------------------------------
// prep kernels: one-time f32 -> bf16 conversions (out of the hot loops)
// ---------------------------------------------------------------------------
__global__ void xprep_kernel(const float* __restrict__ x, bf16* __restrict__ x16) {
    int i = blockIdx.x * blockDim.x + threadIdx.x;       // BATCH*NPTS*32
    if (i < BATCH * NPTS * 32) {
        int c = i & 31, pnt = i >> 5;
        x16[i] = (bf16)((c < 3) ? x[pnt * 3 + c] : 0.f);
    }
}

// Wa[o][c] = W[o][c]; Wb[o][c] = W[o][CIN+c] - W[o][c]  (bf16, padded to CINP)
__global__ void wprep_kernel(const float* __restrict__ W, int CIN, int CINP, int COUT,
                             bf16* __restrict__ Wa, bf16* __restrict__ Wb) {
    int i = blockIdx.x * blockDim.x + threadIdx.x;
    if (i < COUT * CINP) {
        int o = i / CINP, c = i % CINP;
        float a = 0.f, bv = 0.f;
        if (c < CIN) {
            a  = W[(size_t)o * 2 * CIN + c];
            bv = W[(size_t)o * 2 * CIN + CIN + c] - a;
        }
        Wa[i] = (bf16)a;
        Wb[i] = (bf16)bv;
    }
}

__global__ void w5prep_kernel(const float* __restrict__ W5, bf16* __restrict__ W516) {
    int i = blockIdx.x * blockDim.x + threadIdx.x;
    if (i < 512 * 512) W516[i] = (bf16)W5[i];
}

// squared norms from bf16 features (padded channels are zero)
__global__ void norm_kernel(const bf16* __restrict__ F, int stride, int KP,
                            float* __restrict__ norms) {
    int i = blockIdx.x * blockDim.x + threadIdx.x;
    if (i < BATCH * NPTS) {
        const bf16* p = F + (size_t)i * stride;
        float s = 0.f;
        #pragma unroll 1
        for (int c = 0; c < KP; ++c) { float v = (float)p[c]; s += v * v; }
        norms[i] = s;
    }
}

// ---------------------------------------------------------------------------
// kNN: Gram tile via WMMA on pre-converted bf16; register top-20 insertion.
// Scan loops kept rolled (unroll 1) so the insertion code appears once.
// ---------------------------------------------------------------------------
template<int KP>
__global__ __launch_bounds__(32)
void knn_kernel(const bf16* __restrict__ F, int stride,
                const float* __restrict__ norms, int* __restrict__ idxOut) {
    const int rowTile = blockIdx.x;
    const int b       = blockIdx.y;
    const int lane    = threadIdx.x;
    const int g       = lane >> 4, lm = lane & 15;
    const int rowBase = rowTile * 16;
    const bf16*  Fb = F + (size_t)b * NPTS * stride;
    const float* nb = norms + (size_t)b * NPTS;

    constexpr int KS = KP / 32;
    const bf16* arow = Fb + (size_t)(rowBase + lm) * stride;
    v16bf afrag[KS];
    #pragma unroll
    for (int s = 0; s < KS; ++s)
        afrag[s] = load_frag(arow + s * 32 + 8 * g, 16);

    const float rowNorm = nb[rowBase + lm];

    __shared__ float sS[256];
    float bestV[KNN]; int bestI[KNN];
    #pragma unroll
    for (int q = 0; q < KNN; ++q) { bestV[q] = -3.4e38f; bestI[q] = 0; }

    #pragma unroll 1
    for (int ct = 0; ct < NPTS / 16; ++ct) {
        const int colBase = ct * 16;
        const bf16* crow = Fb + (size_t)(colBase + lm) * stride;
        v8f acc = {};
        #pragma unroll
        for (int s = 0; s < KS; ++s)
            acc = wmma_bf16(afrag[s], load_frag(crow + s * 32 + 8 * g, 16), acc);

        #pragma unroll
        for (int r = 0; r < 8; ++r)                 // D layout: M=r+8g, N=lm
            sS[(r + 8 * g) * 16 + lm] = acc[r];
        __syncthreads();
        if (lane < 16) {
            #pragma unroll 1
            for (int nn = 0; nn < 16; ++nn) {
                int col = colBase + nn;
                float negd = 2.0f * sS[lane * 16 + nn] - rowNorm - nb[col];
                if (negd > bestV[KNN - 1]) {
                    float nv = negd; int ni = col;
                    #pragma unroll
                    for (int q = 0; q < KNN; ++q) {
                        if (nv > bestV[q]) {
                            float tv = bestV[q]; int ti = bestI[q];
                            bestV[q] = nv; bestI[q] = ni;
                            nv = tv; ni = ti;
                        }
                    }
                }
            }
        }
        __syncthreads();
    }
    if (lane < 16) {
        int* op = idxOut + ((size_t)b * NPTS + rowBase + lane) * KNN;
        #pragma unroll
        for (int q = 0; q < KNN; ++q) op[q] = bestI[q];
    }
}

// ---------------------------------------------------------------------------
// Dense GEMM: Y[b,n,o] = sum_c A[b,n,c] * Wt[o,c]; wave = 16n x 64o tile.
// Optional fused BN + LeakyReLU (final 512x512 projection).
// ---------------------------------------------------------------------------
template<int KP, bool BN>
__global__ __launch_bounds__(32)
void gemm_kernel(const bf16* __restrict__ A, int aStride,
                 const bf16* __restrict__ Wt,
                 const float* __restrict__ gam, const float* __restrict__ bet,
                 float* __restrict__ Y, int cout) {
    const int nBase = blockIdx.x * 16;
    const int oBase = blockIdx.y * 64;
    const int b     = blockIdx.z;
    const int lane  = threadIdx.x;
    const int g     = lane >> 4, lm = lane & 15;
    const bf16* arow = A + ((size_t)b * NPTS + nBase + lm) * aStride;
    constexpr int KS = KP / 32;

    v8f acc[4] = {v8f{}, v8f{}, v8f{}, v8f{}};
    #pragma unroll
    for (int s = 0; s < KS; ++s) {
        v16bf af = load_frag(arow + s * 32 + 8 * g, 16);
        if (s + 1 < KS) __builtin_prefetch(arow + (s + 1) * 32, 0, 1);
        #pragma unroll
        for (int u = 0; u < 4; ++u) {
            const bf16* brow = Wt + (size_t)(oBase + u * 16 + lm) * KP + s * 32 + 8 * g;
            acc[u] = wmma_bf16(af, load_frag(brow, 16), acc[u]);
        }
    }
    const float rs = rsqrtf(1.0f + 1e-5f);
    #pragma unroll
    for (int u = 0; u < 4; ++u) {
        const int o = oBase + u * 16 + lm;
        float sc = 1.f, bi = 0.f;
        if constexpr (BN) { sc = gam[o] * rs; bi = bet[o]; }
        #pragma unroll
        for (int r = 0; r < 8; ++r) {
            const int m = r + 8 * g;
            float h = acc[u][r];
            if constexpr (BN) { h = h * sc + bi; h = (h >= 0.f) ? h : SLOPE * h; }
            Y[((size_t)b * NPTS + nBase + m) * cout + o] = h;
        }
    }
}

// ---------------------------------------------------------------------------
// EdgeConv epilogue: x_out[n,o] = max_k lrelu(sc*(Q[idx[n,k],o] + P[n,o]) + b)
// Writes bf16 straight into its hcat channel slice.
// ---------------------------------------------------------------------------
__global__ void aggregate_kernel(const float* __restrict__ P, const float* __restrict__ Q,
                                 const int* __restrict__ idx,
                                 const float* __restrict__ gam, const float* __restrict__ bet,
                                 bf16* __restrict__ out, int outStride, int cout) {
    const int n = blockIdx.x, b = blockIdx.y;
    const int o = threadIdx.x;                       // blockDim == cout
    __shared__ int sIdx[KNN];
    if (o < KNN) sIdx[o] = idx[((size_t)b * NPTS + n) * KNN + o];
    __syncthreads();
    const float rs = rsqrtf(1.0f + 1e-5f);
    const size_t base = (size_t)b * NPTS;
    const float p  = P[(base + n) * cout + o];
    const float sc = gam[o] * rs, bi = bet[o];
    float m = -3.4e38f;
    #pragma unroll
    for (int k = 0; k < KNN; ++k) {
        float h = (Q[(base + sIdx[k]) * cout + o] + p) * sc + bi;
        h = (h >= 0.f) ? h : SLOPE * h;
        m = fmaxf(m, h);
    }
    out[(base + n) * outStride + o] = (bf16)m;
}

// max over n -> feat (8,512)
__global__ void maxn_kernel(const float* __restrict__ H, float* __restrict__ outFeat) {
    int i = blockIdx.x * blockDim.x + threadIdx.x;
    if (i < BATCH * 512) {
        int b = i >> 9, o = i & 511;
        const float* p = H + (size_t)b * NPTS * 512 + o;
        float m = -3.4e38f;
        #pragma unroll 1
        for (int n = 0; n < NPTS; ++n) m = fmaxf(m, p[(size_t)n * 512]);
        outFeat[i] = m;
    }
}

// embedding = feat @ Wemb^T (8x512 * 512x256)
__global__ void emb_kernel(const float* __restrict__ feat,
                           const float* __restrict__ Wemb,
                           float* __restrict__ outEmb) {
    int i = blockIdx.x * blockDim.x + threadIdx.x;
    if (i < BATCH * 256) {
        int b = i >> 8, f = i & 255;
        const float* h = feat + (size_t)b * 512;
        const float* w = Wemb + (size_t)f * 512;
        float s = 0.f;
        #pragma unroll 1
        for (int c = 0; c < 512; ++c) s += h[c] * w[c];
        outEmb[i] = s;
    }
}

// ---------------------------------------------------------------------------
extern "C" void kernel_launch(void* const* d_in, const int* in_sizes, int n_in,
                              void* d_out, int out_size, void* d_ws, size_t ws_size,
                              hipStream_t stream) {
    (void)in_sizes; (void)n_in; (void)out_size; (void)ws_size;
    const float* x    = (const float*)d_in[0];
    const float* W1   = (const float*)d_in[1];
    const float* g1   = (const float*)d_in[2];
    const float* b1   = (const float*)d_in[3];
    const float* W2   = (const float*)d_in[4];
    const float* g2   = (const float*)d_in[5];
    const float* b2   = (const float*)d_in[6];
    const float* W3   = (const float*)d_in[7];
    const float* g3   = (const float*)d_in[8];
    const float* b3   = (const float*)d_in[9];
    const float* W4   = (const float*)d_in[10];
    const float* g4   = (const float*)d_in[11];
    const float* b4   = (const float*)d_in[12];
    const float* W5   = (const float*)d_in[13];
    const float* g5   = (const float*)d_in[14];
    const float* b5   = (const float*)d_in[15];
    const float* Wemb = (const float*)d_in[16];

    char* ws = (char*)d_ws;
    size_t off = 0;
    auto alloc = [&](size_t bytes) -> void* {
        void* p = ws + off;
        off += (bytes + 255) & ~(size_t)255;
        return p;
    };
    bf16*  x16   = (bf16*) alloc((size_t)BATCH * NPTS * 32 * 2);
    bf16*  hcat  = (bf16*) alloc((size_t)BATCH * NPTS * 512 * 2);
    int*   idxb  = (int*)  alloc((size_t)BATCH * NPTS * KNN * 4);
    float* norms = (float*)alloc((size_t)BATCH * NPTS * 4);
    float* Pbuf  = (float*)alloc((size_t)BATCH * NPTS * 256 * 4);
    float* Qbuf  = (float*)alloc((size_t)BATCH * NPTS * 256 * 4);
    bf16*  Wa16  = (bf16*) alloc((size_t)256 * 128 * 2);
    bf16*  Wb16  = (bf16*) alloc((size_t)256 * 128 * 2);
    bf16*  W516  = (bf16*) alloc((size_t)512 * 512 * 2);
    float* Hbuf  = (float*)alloc((size_t)BATCH * NPTS * 512 * 4);

    const dim3 wv(32);
    const dim3 gKnn(NPTS / 16, BATCH);
    const int  nPt = (BATCH * NPTS + 255) / 256;

    xprep_kernel<<<(BATCH * NPTS * 32 + 255) / 256, 256, 0, stream>>>(x, x16);
    w5prep_kernel<<<(512 * 512 + 255) / 256, 256, 0, stream>>>(W5, W516);

    // ---- layer 1 : 3 -> 64 (padded K=32) ----
    wprep_kernel<<<(64 * 32 + 255) / 256, 256, 0, stream>>>(W1, 3, 32, 64, Wa16, Wb16);
    norm_kernel<<<nPt, 256, 0, stream>>>(x16, 32, 32, norms);
    knn_kernel<32><<<gKnn, wv, 0, stream>>>(x16, 32, norms, idxb);
    gemm_kernel<32, false><<<dim3(NPTS / 16, 1, BATCH), wv, 0, stream>>>(
        x16, 32, Wa16, nullptr, nullptr, Qbuf, 64);
    gemm_kernel<32, false><<<dim3(NPTS / 16, 1, BATCH), wv, 0, stream>>>(
        x16, 32, Wb16, nullptr, nullptr, Pbuf, 64);
    aggregate_kernel<<<dim3(NPTS, BATCH), 64, 0, stream>>>(
        Pbuf, Qbuf, idxb, g1, b1, hcat + 0, 512, 64);

    // ---- layer 2 : 64 -> 64 ----
    wprep_kernel<<<(64 * 64 + 255) / 256, 256, 0, stream>>>(W2, 64, 64, 64, Wa16, Wb16);
    norm_kernel<<<nPt, 256, 0, stream>>>(hcat + 0, 512, 64, norms);
    knn_kernel<64><<<gKnn, wv, 0, stream>>>(hcat + 0, 512, norms, idxb);
    gemm_kernel<64, false><<<dim3(NPTS / 16, 1, BATCH), wv, 0, stream>>>(
        hcat + 0, 512, Wa16, nullptr, nullptr, Qbuf, 64);
    gemm_kernel<64, false><<<dim3(NPTS / 16, 1, BATCH), wv, 0, stream>>>(
        hcat + 0, 512, Wb16, nullptr, nullptr, Pbuf, 64);
    aggregate_kernel<<<dim3(NPTS, BATCH), 64, 0, stream>>>(
        Pbuf, Qbuf, idxb, g2, b2, hcat + 64, 512, 64);

    // ---- layer 3 : 64 -> 128 ----
    wprep_kernel<<<(128 * 64 + 255) / 256, 256, 0, stream>>>(W3, 64, 64, 128, Wa16, Wb16);
    norm_kernel<<<nPt, 256, 0, stream>>>(hcat + 64, 512, 64, norms);
    knn_kernel<64><<<gKnn, wv, 0, stream>>>(hcat + 64, 512, norms, idxb);
    gemm_kernel<64, false><<<dim3(NPTS / 16, 2, BATCH), wv, 0, stream>>>(
        hcat + 64, 512, Wa16, nullptr, nullptr, Qbuf, 128);
    gemm_kernel<64, false><<<dim3(NPTS / 16, 2, BATCH), wv, 0, stream>>>(
        hcat + 64, 512, Wb16, nullptr, nullptr, Pbuf, 128);
    aggregate_kernel<<<dim3(NPTS, BATCH), 128, 0, stream>>>(
        Pbuf, Qbuf, idxb, g3, b3, hcat + 128, 512, 128);

    // ---- layer 4 : 128 -> 256 ----
    wprep_kernel<<<(256 * 128 + 255) / 256, 256, 0, stream>>>(W4, 128, 128, 256, Wa16, Wb16);
    norm_kernel<<<nPt, 256, 0, stream>>>(hcat + 128, 512, 128, norms);
    knn_kernel<128><<<gKnn, wv, 0, stream>>>(hcat + 128, 512, norms, idxb);
    gemm_kernel<128, false><<<dim3(NPTS / 16, 4, BATCH), wv, 0, stream>>>(
        hcat + 128, 512, Wa16, nullptr, nullptr, Qbuf, 256);
    gemm_kernel<128, false><<<dim3(NPTS / 16, 4, BATCH), wv, 0, stream>>>(
        hcat + 128, 512, Wb16, nullptr, nullptr, Pbuf, 256);
    aggregate_kernel<<<dim3(NPTS, BATCH), 256, 0, stream>>>(
        Pbuf, Qbuf, idxb, g4, b4, hcat + 256, 512, 256);

    // ---- final projection + pooling + embedding ----
    gemm_kernel<512, true><<<dim3(NPTS / 16, 8, BATCH), wv, 0, stream>>>(
        hcat, 512, W516, g5, b5, Hbuf, 512);
    float* outF = (float*)d_out;          // feat (8,1,512)
    float* outE = outF + BATCH * 512;     // embedding (8,256)
    maxn_kernel<<<(BATCH * 512 + 255) / 256, 256, 0, stream>>>(Hbuf, outF);
    emb_kernel<<<(BATCH * 256 + 255) / 256, 256, 0, stream>>>(outF, Wemb, outE);
}
